// MyExperts_20444044329634
// MI455X (gfx1250) — matrix operand order repaired and checked by reading, not compiled, not run on hardware
//
#include <hip/hip_runtime.h>
#include <hip/hip_bf16.h>
#include <cstdint>

// MoE grouped FFN for MI455X (gfx1250, wave32, WMMA).
// Preferred path (needs ~49MB ws per expert):
//   pass 0: per-expert weight convert+transpose fp32[K,N] -> bf16[N,K]  (ws)
//   pass 1: A = silu(X*W1+b1)*(X*W2+b2), bf16 -> ws   (B tiles via async DMA)
//   pass 2: out = (A*W3+b3)*probs                     (A & B tiles via async DMA)
// All GEMMs: v_wmma_f32_16x16x32_bf16, fp32 accumulate, double-buffered LDS.
// Fallback path (small ws): in-kernel fp32->bf16 conversion (single-buffered).

typedef __bf16 bf16_t;
typedef __attribute__((ext_vector_type(16))) __bf16 v16bf;
typedef __attribute__((ext_vector_type(8)))  __bf16 v8bf;
typedef __attribute__((ext_vector_type(8)))  float  v8f;

#define BM   128
#define BN   128
#define BK   32
#define KPAD 16
#define LDK  (BK + KPAD)   // LDS row stride: 96B, 16B-aligned, bank-skewed

// ---- WMMA fragment loads from LDS --------------------------------------
// A frag (16x32): lane(lo,hi) row M=lo; elem j: K = hi*8 + (j&7) + (j>=8)*16
static __device__ __forceinline__ v16bf load_frag_a(const bf16_t* prow, int hi) {
    v8bf c0 = *(const v8bf*)(prow + hi * 8);
    v8bf c1 = *(const v8bf*)(prow + 16 + hi * 8);
    v16bf a;
#pragma unroll
    for (int i = 0; i < 8; ++i) { a[i] = c0[i]; a[i + 8] = c1[i]; }
    return a;
}
// B frag (32x16): lane(lo,hi) col N=lo; elem j: K = hi*8 + (j&7) + (j>=8)*16  (LDS is [N][K])
static __device__ __forceinline__ v16bf load_frag_b(const bf16_t* pcol, int hi) {
    v8bf c0 = *(const v8bf*)(pcol + hi * 8);
    v8bf c1 = *(const v8bf*)(pcol + 16 + hi * 8);
    v16bf b;
#pragma unroll
    for (int i = 0; i < 8; ++i) { b[i] = c0[i]; b[i + 8] = c1[i]; }
    return b;
}

// CDNA5 async 16B global->LDS copy (GVS mode), tracked by ASYNCcnt.
static __device__ __forceinline__ void async_copy_b128(void* lds_ptr,
                                                       const void* gbase,
                                                       uint32_t goff_bytes) {
    uint32_t lds = (uint32_t)(uintptr_t)lds_ptr;  // low 32 bits = LDS offset
    asm volatile("global_load_async_to_lds_b128 %0, %1, %2 offset:0"
                 :: "v"(lds), "v"(goff_bytes), "s"(gbase)
                 : "memory");
}
static __device__ __forceinline__ void wait_asynccnt0() {
    asm volatile("s_wait_asynccnt 0x0" ::: "memory");
}

// Stage a [128 rows x 32 K] bf16 tile (global [N,K] layout) into LDS via async DMA.
// 2 x b128 per thread (256 threads): 128 rows x 4 chunks of 8 bf16.
static __device__ __forceinline__ void issue_tile_async(bf16_t (*Ls)[LDK],
                                                        const bf16_t* gbase,
                                                        int tid, int row0, int k0, int Kd) {
#pragma unroll
    for (int it = 0; it < 2; ++it) {
        int u = tid + it * 256;
        int r = u >> 2, c = (u & 3) << 3;
        uint32_t go = (uint32_t)(((size_t)(row0 + r) * Kd + (size_t)(k0 + c)) * sizeof(bf16_t));
        async_copy_b128(&Ls[r][c], gbase, go);
    }
}

// ---- Pass 0: weight convert + transpose: fp32 [K,N] -> bf16 [N,K] ------
__global__ __launch_bounds__(256)
void convert_transpose_kernel(const float* __restrict__ src,  // [K, N]
                              bf16_t* __restrict__ dst,       // [N, K]
                              int K, int N) {
    __shared__ float tile[32][33];
    const int n0 = blockIdx.x * 32;
    const int k0 = blockIdx.y * 32;
    const int tx = threadIdx.x & 31;
    const int ty = threadIdx.x >> 5;   // 0..7
#pragma unroll
    for (int i = 0; i < 4; ++i) {
        int r = ty + i * 8;
        tile[r][tx] = src[(size_t)(k0 + r) * N + (n0 + tx)];
    }
    __syncthreads();
#pragma unroll
    for (int i = 0; i < 4; ++i) {
        int r = ty + i * 8;
        dst[(size_t)(n0 + r) * K + (k0 + tx)] = (bf16_t)tile[tx][r];
    }
}

// ---- Pass 1 (preferred): gate/up with pre-transposed bf16 weights ------
__global__ __launch_bounds__(256)
void moe_gate_up_tw_kernel(const float*  __restrict__ X,    // [rows, H] fp32
                           const bf16_t* __restrict__ W1t,  // [F, H] bf16
                           const bf16_t* __restrict__ W2t,  // [F, H] bf16
                           const float*  __restrict__ b1,   // [F]
                           const float*  __restrict__ b2,   // [F]
                           bf16_t* __restrict__ Aout,       // [rows, F] bf16
                           int H, int F) {
    __shared__ bf16_t Xs [2][BM][LDK];
    __shared__ bf16_t W1s[2][BN][LDK];
    __shared__ bf16_t W2s[2][BN][LDK];

    const int tid  = threadIdx.x;
    const int lane = tid & 31;
    const int wave = tid >> 5;
    const int lo   = lane & 15;
    const int hi   = lane >> 4;
    const int wm   = (wave & 1) * 64;
    const int wn   = (wave >> 1) * 32;

    const int m0 = blockIdx.y * BM;
    const int n0 = blockIdx.x * BN;

    const v8f vzero = {0.f, 0.f, 0.f, 0.f, 0.f, 0.f, 0.f, 0.f};
    v8f acc_g[4][2], acc_u[4][2];
#pragma unroll
    for (int mt = 0; mt < 4; ++mt)
#pragma unroll
        for (int nt = 0; nt < 2; ++nt) { acc_g[mt][nt] = vzero; acc_u[mt][nt] = vzero; }

    const int nk = H / BK;
    float4 fx[4];

    // ---- prologue: stage tile 0 into buffer 0
    issue_tile_async(W1s[0], W1t, tid, n0, 0, H);
    issue_tile_async(W2s[0], W2t, tid, n0, 0, H);
#pragma unroll
    for (int it = 0; it < 4; ++it) {
        int v = tid + it * 256;
        fx[it] = *(const float4*)(X + (size_t)(m0 + (v >> 3)) * H + ((v & 7) << 2));
    }
#pragma unroll
    for (int it = 0; it < 4; ++it) {
        int v = tid + it * 256;
        bf16_t* d = &Xs[0][v >> 3][(v & 7) << 2];
        d[0] = (bf16_t)fx[it].x; d[1] = (bf16_t)fx[it].y;
        d[2] = (bf16_t)fx[it].z; d[3] = (bf16_t)fx[it].w;
    }
    wait_asynccnt0();

    for (int kt = 0; kt < nk; ++kt) {
        __syncthreads();                    // buf[kt&1] ready for everyone
        const int cur = kt & 1, nxt = cur ^ 1;
        const bool more = (kt + 1 < nk);
        if (more) {
            const int kn = (kt + 1) * BK;
            issue_tile_async(W1s[nxt], W1t, tid, n0, kn, H);
            issue_tile_async(W2s[nxt], W2t, tid, n0, kn, H);
#pragma unroll
            for (int it = 0; it < 4; ++it) {
                int v = tid + it * 256;
                fx[it] = *(const float4*)(X + (size_t)(m0 + (v >> 3)) * H + (kn + ((v & 7) << 2)));
            }
        }
        // ---- compute on current buffer (overlaps in-flight fetches)
        v16bf af[4], bg[2], bu[2];
#pragma unroll
        for (int mt = 0; mt < 4; ++mt)
            af[mt] = load_frag_a(&Xs[cur][wm + mt * 16 + lo][0], hi);
#pragma unroll
        for (int nt = 0; nt < 2; ++nt) {
            bg[nt] = load_frag_b(&W1s[cur][wn + nt * 16 + lo][0], hi);
            bu[nt] = load_frag_b(&W2s[cur][wn + nt * 16 + lo][0], hi);
        }
#pragma unroll
        for (int mt = 0; mt < 4; ++mt)
#pragma unroll
            for (int nt = 0; nt < 2; ++nt) {
                acc_g[mt][nt] = __builtin_amdgcn_wmma_f32_16x16x32_bf16(
                    false, af[mt], false, bg[nt], (short)0, acc_g[mt][nt], false, false);
                acc_u[mt][nt] = __builtin_amdgcn_wmma_f32_16x16x32_bf16(
                    false, af[mt], false, bu[nt], (short)0, acc_u[mt][nt], false, false);
            }
        if (more) {
#pragma unroll
            for (int it = 0; it < 4; ++it) {
                int v = tid + it * 256;
                bf16_t* d = &Xs[nxt][v >> 3][(v & 7) << 2];
                d[0] = (bf16_t)fx[it].x; d[1] = (bf16_t)fx[it].y;
                d[2] = (bf16_t)fx[it].z; d[3] = (bf16_t)fx[it].w;
            }
        }
        wait_asynccnt0();                   // next buffer's DMA landed
    }

    // Epilogue: silu(gate)*up -> bf16. D layout: M=i+8*hi, N=lo.
#pragma unroll
    for (int mt = 0; mt < 4; ++mt)
#pragma unroll
        for (int nt = 0; nt < 2; ++nt) {
            const int gn = n0 + wn + nt * 16 + lo;
            const float bb1 = b1[gn];
            const float bb2 = b2[gn];
#pragma unroll
            for (int i = 0; i < 8; ++i) {
                const int gm = m0 + wm + mt * 16 + hi * 8 + i;
                const float g = acc_g[mt][nt][i] + bb1;
                const float u = acc_u[mt][nt][i] + bb2;
                const float s = g / (1.0f + __expf(-g));
                Aout[(size_t)gm * F + gn] = (bf16_t)(s * u);
            }
        }
}

// ---- Pass 2 (preferred): down-proj, fully async-staged -----------------
__global__ __launch_bounds__(256)
void moe_down_tw_kernel(const bf16_t* __restrict__ Ain,   // [rows, F] bf16
                        const bf16_t* __restrict__ W3t,   // [H, F] bf16
                        const float*  __restrict__ b3,    // [H]
                        const float*  __restrict__ probs, // [rows]
                        float* __restrict__ out,          // [rows, H]
                        int F, int H) {
    __shared__ bf16_t As [2][BM][LDK];
    __shared__ bf16_t W3s[2][BN][LDK];

    const int tid  = threadIdx.x;
    const int lane = tid & 31;
    const int wave = tid >> 5;
    const int lo   = lane & 15;
    const int hi   = lane >> 4;
    const int wm   = (wave & 1) * 64;
    const int wn   = (wave >> 1) * 32;

    const int m0 = blockIdx.y * BM;
    const int n0 = blockIdx.x * BN;

    const v8f vzero = {0.f, 0.f, 0.f, 0.f, 0.f, 0.f, 0.f, 0.f};
    v8f acc[4][2];
#pragma unroll
    for (int mt = 0; mt < 4; ++mt)
#pragma unroll
        for (int nt = 0; nt < 2; ++nt) acc[mt][nt] = vzero;

    const int nk = F / BK;

    issue_tile_async(As[0],  Ain, tid, m0, 0, F);
    issue_tile_async(W3s[0], W3t, tid, n0, 0, F);
    wait_asynccnt0();

    for (int kt = 0; kt < nk; ++kt) {
        __syncthreads();
        const int cur = kt & 1, nxt = cur ^ 1;
        if (kt + 1 < nk) {
            const int kn = (kt + 1) * BK;
            issue_tile_async(As[nxt],  Ain, tid, m0, kn, F);
            issue_tile_async(W3s[nxt], W3t, tid, n0, kn, F);
        }
        v16bf af[4], bf[2];
#pragma unroll
        for (int mt = 0; mt < 4; ++mt)
            af[mt] = load_frag_a(&As[cur][wm + mt * 16 + lo][0], hi);
#pragma unroll
        for (int nt = 0; nt < 2; ++nt)
            bf[nt] = load_frag_b(&W3s[cur][wn + nt * 16 + lo][0], hi);
#pragma unroll
        for (int mt = 0; mt < 4; ++mt)
#pragma unroll
            for (int nt = 0; nt < 2; ++nt)
                acc[mt][nt] = __builtin_amdgcn_wmma_f32_16x16x32_bf16(
                    false, af[mt], false, bf[nt], (short)0, acc[mt][nt], false, false);
        wait_asynccnt0();
    }

#pragma unroll
    for (int mt = 0; mt < 4; ++mt)
#pragma unroll
        for (int nt = 0; nt < 2; ++nt) {
            const int gn = n0 + wn + nt * 16 + lo;
            const float bb = b3[gn];
#pragma unroll
            for (int i = 0; i < 8; ++i) {
                const int gm = m0 + wm + mt * 16 + hi * 8 + i;
                out[(size_t)gm * H + gn] = (acc[mt][nt][i] + bb) * probs[gm];
            }
        }
}

// ---- Fallback kernels (small workspace): in-kernel conversion ----------
__global__ __launch_bounds__(256)
void moe_gate_up_fb_kernel(const float* __restrict__ X, const float* __restrict__ W1,
                           const float* __restrict__ W2, const float* __restrict__ b1,
                           const float* __restrict__ b2, bf16_t* __restrict__ Aout,
                           int H, int F) {
    __shared__ bf16_t Xs [BM][LDK];
    __shared__ bf16_t W1s[BN][LDK];
    __shared__ bf16_t W2s[BN][LDK];
    const int tid = threadIdx.x, lane = tid & 31, wave = tid >> 5;
    const int lo = lane & 15, hi = lane >> 4;
    const int wm = (wave & 1) * 64, wn = (wave >> 1) * 32;
    const int m0 = blockIdx.y * BM, n0 = blockIdx.x * BN;
    const v8f vzero = {0.f, 0.f, 0.f, 0.f, 0.f, 0.f, 0.f, 0.f};
    v8f acc_g[4][2], acc_u[4][2];
#pragma unroll
    for (int mt = 0; mt < 4; ++mt)
#pragma unroll
        for (int nt = 0; nt < 2; ++nt) { acc_g[mt][nt] = vzero; acc_u[mt][nt] = vzero; }
    const int nk = H / BK;
    for (int kt = 0; kt < nk; ++kt) {
        const int k0 = kt * BK;
        __syncthreads();
        float4 fx[4], f1[4], f2[4];
#pragma unroll
        for (int it = 0; it < 4; ++it) {
            int v = tid + it * 256;
            fx[it] = *(const float4*)(X + (size_t)(m0 + (v >> 3)) * H + (k0 + ((v & 7) << 2)));
        }
#pragma unroll
        for (int it = 0; it < 4; ++it) {
            int v = tid + it * 256;
            size_t go = (size_t)(k0 + (v >> 5)) * F + (n0 + ((v & 31) << 2));
            f1[it] = *(const float4*)(W1 + go);
            f2[it] = *(const float4*)(W2 + go);
        }
#pragma unroll
        for (int it = 0; it < 4; ++it) {
            int v = tid + it * 256;
            bf16_t* d = &Xs[v >> 3][(v & 7) << 2];
            d[0] = (bf16_t)fx[it].x; d[1] = (bf16_t)fx[it].y;
            d[2] = (bf16_t)fx[it].z; d[3] = (bf16_t)fx[it].w;
        }
#pragma unroll
        for (int it = 0; it < 4; ++it) {
            int v = tid + it * 256;
            int kr = v >> 5, nc = (v & 31) << 2;
            W1s[nc + 0][kr] = (bf16_t)f1[it].x; W1s[nc + 1][kr] = (bf16_t)f1[it].y;
            W1s[nc + 2][kr] = (bf16_t)f1[it].z; W1s[nc + 3][kr] = (bf16_t)f1[it].w;
            W2s[nc + 0][kr] = (bf16_t)f2[it].x; W2s[nc + 1][kr] = (bf16_t)f2[it].y;
            W2s[nc + 2][kr] = (bf16_t)f2[it].z; W2s[nc + 3][kr] = (bf16_t)f2[it].w;
        }
        __syncthreads();
        v16bf af[4], bg[2], bu[2];
#pragma unroll
        for (int mt = 0; mt < 4; ++mt)
            af[mt] = load_frag_a(&Xs[wm + mt * 16 + lo][0], hi);
#pragma unroll
        for (int nt = 0; nt < 2; ++nt) {
            bg[nt] = load_frag_b(&W1s[wn + nt * 16 + lo][0], hi);
            bu[nt] = load_frag_b(&W2s[wn + nt * 16 + lo][0], hi);
        }
#pragma unroll
        for (int mt = 0; mt < 4; ++mt)
#pragma unroll
            for (int nt = 0; nt < 2; ++nt) {
                acc_g[mt][nt] = __builtin_amdgcn_wmma_f32_16x16x32_bf16(
                    false, af[mt], false, bg[nt], (short)0, acc_g[mt][nt], false, false);
                acc_u[mt][nt] = __builtin_amdgcn_wmma_f32_16x16x32_bf16(
                    false, af[mt], false, bu[nt], (short)0, acc_u[mt][nt], false, false);
            }
    }
#pragma unroll
    for (int mt = 0; mt < 4; ++mt)
#pragma unroll
        for (int nt = 0; nt < 2; ++nt) {
            const int gn = n0 + wn + nt * 16 + lo;
            const float bb1 = b1[gn], bb2 = b2[gn];
#pragma unroll
            for (int i = 0; i < 8; ++i) {
                const int gm = m0 + wm + mt * 16 + hi * 8 + i;
                const float g = acc_g[mt][nt][i] + bb1;
                const float u = acc_u[mt][nt][i] + bb2;
                const float s = g / (1.0f + __expf(-g));
                Aout[(size_t)gm * F + gn] = (bf16_t)(s * u);
            }
        }
}

__global__ __launch_bounds__(256)
void moe_down_fb_kernel(const bf16_t* __restrict__ Ain, const float* __restrict__ W3,
                        const float* __restrict__ b3, const float* __restrict__ probs,
                        float* __restrict__ out, int F, int H) {
    __shared__ bf16_t As [BM][LDK];
    __shared__ bf16_t W3s[BN][LDK];
    const int tid = threadIdx.x, lane = tid & 31, wave = tid >> 5;
    const int lo = lane & 15, hi = lane >> 4;
    const int wm = (wave & 1) * 64, wn = (wave >> 1) * 32;
    const int m0 = blockIdx.y * BM, n0 = blockIdx.x * BN;
    const v8f vzero = {0.f, 0.f, 0.f, 0.f, 0.f, 0.f, 0.f, 0.f};
    v8f acc[4][2];
#pragma unroll
    for (int mt = 0; mt < 4; ++mt)
#pragma unroll
        for (int nt = 0; nt < 2; ++nt) acc[mt][nt] = vzero;
    const int nk = F / BK;
    for (int kt = 0; kt < nk; ++kt) {
        const int k0 = kt * BK;
        __syncthreads();
        issue_tile_async(As, Ain, tid, m0, k0, F);
        float4 f3[4];
#pragma unroll
        for (int it = 0; it < 4; ++it) {
            int v = tid + it * 256;
            f3[it] = *(const float4*)(W3 + (size_t)(k0 + (v >> 5)) * H + (n0 + ((v & 31) << 2)));
        }
#pragma unroll
        for (int it = 0; it < 4; ++it) {
            int v = tid + it * 256;
            int kr = v >> 5, nc = (v & 31) << 2;
            W3s[nc + 0][kr] = (bf16_t)f3[it].x; W3s[nc + 1][kr] = (bf16_t)f3[it].y;
            W3s[nc + 2][kr] = (bf16_t)f3[it].z; W3s[nc + 3][kr] = (bf16_t)f3[it].w;
        }
        wait_asynccnt0();
        __syncthreads();
        v16bf af[4], bf[2];
#pragma unroll
        for (int mt = 0; mt < 4; ++mt)
            af[mt] = load_frag_a(&As[wm + mt * 16 + lo][0], hi);
#pragma unroll
        for (int nt = 0; nt < 2; ++nt)
            bf[nt] = load_frag_b(&W3s[wn + nt * 16 + lo][0], hi);
#pragma unroll
        for (int mt = 0; mt < 4; ++mt)
#pragma unroll
            for (int nt = 0; nt < 2; ++nt)
                acc[mt][nt] = __builtin_amdgcn_wmma_f32_16x16x32_bf16(
                    false, af[mt], false, bf[nt], (short)0, acc[mt][nt], false, false);
    }
#pragma unroll
    for (int mt = 0; mt < 4; ++mt)
#pragma unroll
        for (int nt = 0; nt < 2; ++nt) {
            const int gn = n0 + wn + nt * 16 + lo;
            const float bb = b3[gn];
#pragma unroll
            for (int i = 0; i < 8; ++i) {
                const int gm = m0 + wm + mt * 16 + hi * 8 + i;
                out[(size_t)gm * H + gn] = (acc[mt][nt][i] + bb) * probs[gm];
            }
        }
}

// ---- Host launch --------------------------------------------------------
extern "C" void kernel_launch(void* const* d_in, const int* in_sizes, int n_in,
                              void* d_out, int out_size, void* d_ws, size_t ws_size,
                              hipStream_t stream) {
    const float* X     = (const float*)d_in[0];
    const float* probs = (const float*)d_in[2];
    const float* W1    = (const float*)d_in[3];
    const float* b1    = (const float*)d_in[4];
    const float* W2    = (const float*)d_in[5];
    const float* b2    = (const float*)d_in[6];
    const float* W3    = (const float*)d_in[7];
    const float* b3    = (const float*)d_in[8];
    float*       out   = (float*)d_out;

    const int E = in_sizes[1];
    const int H = in_sizes[8] / E;
    const int F = in_sizes[4] / E;
    const int T = (in_sizes[0] / H) / E;

    const size_t FH      = (size_t)F * H;
    const size_t w_elems = 3 * FH;                       // W1t + W2t + W3t (bf16)
    const size_t need    = (w_elems + (size_t)BM * F) * sizeof(bf16_t);

    if (ws_size >= need) {
        // ---- preferred: pre-converted/transposed bf16 weights + async DMA
        bf16_t* W1t  = (bf16_t*)d_ws;
        bf16_t* W2t  = W1t + FH;
        bf16_t* W3t  = W2t + FH;
        bf16_t* Abuf = W3t + FH;
        long long max_rows =
            (long long)((ws_size / sizeof(bf16_t) - w_elems) / (size_t)F);
        max_rows &= ~(long long)(BM - 1);
        if (max_rows > T) max_rows = T;

        for (int e = 0; e < E; ++e) {
            convert_transpose_kernel<<<dim3(F / 32, H / 32), 256, 0, stream>>>(
                W1 + (size_t)e * FH, W1t, H, F);
            convert_transpose_kernel<<<dim3(F / 32, H / 32), 256, 0, stream>>>(
                W2 + (size_t)e * FH, W2t, H, F);
            convert_transpose_kernel<<<dim3(H / 32, F / 32), 256, 0, stream>>>(
                W3 + (size_t)e * FH, W3t, F, H);
            for (int r0 = 0; r0 < T; r0 += (int)max_rows) {
                const int rows = (T - r0 < (int)max_rows) ? (T - r0) : (int)max_rows;
                moe_gate_up_tw_kernel<<<dim3(F / BN, rows / BM), 256, 0, stream>>>(
                    X + ((size_t)e * T + r0) * H, W1t, W2t,
                    b1 + (size_t)e * F, b2 + (size_t)e * F, Abuf, H, F);
                moe_down_tw_kernel<<<dim3(H / BN, rows / BM), 256, 0, stream>>>(
                    Abuf, W3t, b3 + (size_t)e * H, probs + (size_t)e * T + r0,
                    out + ((size_t)e * T + r0) * H, F, H);
            }
        }
    } else {
        // ---- fallback: in-kernel conversion, A-chunk only workspace
        bf16_t* Abuf = (bf16_t*)d_ws;
        long long max_rows = (long long)(ws_size / ((size_t)F * sizeof(bf16_t)));
        max_rows &= ~(long long)(BM - 1);
        if (max_rows > T)  max_rows = T;
        if (max_rows < BM) max_rows = BM;
        for (int e = 0; e < E; ++e) {
            for (int r0 = 0; r0 < T; r0 += (int)max_rows) {
                const int rows = (T - r0 < (int)max_rows) ? (T - r0) : (int)max_rows;
                moe_gate_up_fb_kernel<<<dim3(F / BN, rows / BM), 256, 0, stream>>>(
                    X + ((size_t)e * T + r0) * H,
                    W1 + (size_t)e * FH, W2 + (size_t)e * FH,
                    b1 + (size_t)e * F, b2 + (size_t)e * F, Abuf, H, F);
                moe_down_fb_kernel<<<dim3(H / BN, rows / BM), 256, 0, stream>>>(
                    Abuf, W3 + (size_t)e * FH, b3 + (size_t)e * H,
                    probs + (size_t)e * T + r0,
                    out + ((size_t)e * T + r0) * H, F, H);
            }
        }
    }
}
